// SelfAttention_43061342110468
// MI455X (gfx1250) — matrix operand order repaired and checked by reading, not compiled, not run on hardware
//
#include <hip/hip_runtime.h>
#include <math.h>

// ---------------------------------------------------------------------------
// Problem constants (from reference): B=4, S=2048, D=512, H=8, Dh=64
// ---------------------------------------------------------------------------
#define BB   4
#define SS   2048
#define DD   512
#define HH   8
#define DH   64
#define MROWS (BB * SS)          // 8192
#define NQKV  (3 * DD)           // 1536
#define SCALE 0.125f             // 1/sqrt(64)

typedef __attribute__((ext_vector_type(16))) __bf16 v16bf;
typedef __attribute__((ext_vector_type(8)))  __bf16 v8bf;
typedef __attribute__((ext_vector_type(8)))  float  v8f;
typedef __attribute__((ext_vector_type(4)))  int    v4i;

__device__ __forceinline__ unsigned short f2bf(float f) {
    unsigned int u = __float_as_uint(f);
    unsigned int r = u + 0x7fffu + ((u >> 16) & 1u);   // round-to-nearest-even
    return (unsigned short)(r >> 16);
}

// ---------------------------------------------------------------------------
// CDNA5 async global->LDS staging (ASYNCcnt path), with portable fallback.
// Host pass / older toolchains take the sync fallback automatically.
// ---------------------------------------------------------------------------
#if __has_builtin(__builtin_amdgcn_global_load_async_to_lds_b128) && \
    __has_builtin(__builtin_amdgcn_s_wait_asynccnt)
#define USE_ASYNC_LDS 1
#else
#define USE_ASYNC_LDS 0
#endif

__device__ __forceinline__ void cp16_g2l(const unsigned short* g, unsigned short* l) {
#if USE_ASYNC_LDS
    __builtin_amdgcn_global_load_async_to_lds_b128(
        (__attribute__((address_space(1))) v4i*)g,
        (__attribute__((address_space(3))) v4i*)l, 0, 0);
#else
    *(uint4*)l = *(const uint4*)g;
#endif
}

__device__ __forceinline__ void cp_join() {
#if USE_ASYNC_LDS
    __builtin_amdgcn_s_wait_asynccnt(0);
#endif
}

__device__ __forceinline__ v8f bwmma(v16bf a, v16bf b, v8f c) {
    return __builtin_amdgcn_wmma_f32_16x16x32_bf16(false, a, false, b,
                                                   (short)0, c, false, false);
}

// ---------------------------------------------------------------------------
// Elementwise helpers
// ---------------------------------------------------------------------------
__global__ __launch_bounds__(256)
void cvt_pack_kernel(const float* __restrict__ src, unsigned short* __restrict__ dst,
                     int rows, int cols) {
    int idx = blockIdx.x * 256 + threadIdx.x;
    if (idx >= rows * cols) return;
    dst[idx] = f2bf(src[idx]);
}

// transpose-convert: dst[(row_off + c)][r] over dst row length `rows`
__global__ __launch_bounds__(256)
void cvt_pack_t_kernel(const float* __restrict__ src, unsigned short* __restrict__ dst,
                       int rows, int cols, int dst_row_off) {
    int idx = blockIdx.x * 256 + threadIdx.x;
    if (idx >= rows * cols) return;
    int r = idx / cols, c = idx % cols;
    dst[(size_t)(dst_row_off + c) * rows + r] = f2bf(src[idx]);
}

__global__ __launch_bounds__(256)
void bias_pack_kernel(const float* __restrict__ bq, const float* __restrict__ bk,
                      const float* __restrict__ bv, float* __restrict__ dst) {
    int idx = blockIdx.x * 256 + threadIdx.x;
    if (idx >= NQKV) return;
    if (idx < DD)            dst[idx] = bq[idx];
    else if (idx < 2 * DD)   dst[idx] = bk[idx - DD];
    else                     dst[idx] = bv[idx - 2 * DD];
}

__global__ __launch_bounds__(256)
void qk_cvt_kernel(const float* __restrict__ qkv,
                   unsigned short* __restrict__ qb, unsigned short* __restrict__ kb) {
    int idx = blockIdx.x * 256 + threadIdx.x;      // over 8192*512
    int row = idx >> 9, col = idx & 511;
    size_t base = (size_t)row * NQKV;
    qb[idx] = f2bf(qkv[base + col]);
    kb[idx] = f2bf(qkv[base + DD + col]);
}

__global__ void zero_s_kernel(float* s) {
    if (threadIdx.x < BB * HH) s[threadIdx.x] = 0.0f;
}

__global__ __launch_bounds__(256)
void scale_v_kernel(const float* __restrict__ qkv, const float* __restrict__ s,
                    unsigned short* __restrict__ att) {
    int idx = blockIdx.x * 256 + threadIdx.x;      // over 8192*512
    int row = idx >> 9, col = idx & 511;
    int b = row >> 11, h = col >> 6;
    float v = qkv[(size_t)row * NQKV + 2 * DD + col] * s[b * HH + h];
    att[idx] = f2bf(v);
}

// ---------------------------------------------------------------------------
// bf16 WMMA GEMM: C[M,N] = A[M,K] * Bt[N,K]^T + bias[N]   (A,Bt bf16; C f32)
// Block 256 = 8 waves (2x4); block tile 64x128; wave tile 32x32 (4 WMMA/kstep).
// Both operands staged with contiguous 16B copies (async-to-LDS on CDNA5).
// ---------------------------------------------------------------------------
__global__ __launch_bounds__(256)
void gemm_bf16_kernel(const unsigned short* __restrict__ A,   // M x K
                      const unsigned short* __restrict__ Bt,  // N x K (pre-transposed)
                      const float* __restrict__ bias,         // N
                      float* __restrict__ C,                  // M x N
                      int M, int N, int K) {
    __shared__ unsigned short As[64][40];    // [m][k] + pad
    __shared__ unsigned short Bs[128][40];   // [n][k] + pad

    const int tid  = threadIdx.x;
    const int wave = tid >> 5;
    const int lane = tid & 31;
    const int r    = lane & 15;
    const int hi   = (lane >> 4) & 1;
    const int wy   = wave >> 2;              // 0..1  (m)
    const int wx   = wave & 3;               // 0..3  (n)

    const int m0 = blockIdx.x * 64;
    const int n0 = blockIdx.y * 128;

    // staging coordinates: A 64x32 = 256 lanes x 8 elems; Bt 128x32 = 256 x 16
    const int arow = (tid * 8) >> 5;         // 0..63
    const int acol = (tid * 8) & 31;         // 0,8,16,24
    const int brow = tid >> 1;               // 0..127
    const int bcol = (tid & 1) * 16;         // 0 or 16

    const unsigned short* aG = A  + (size_t)(m0 + arow) * K + acol;
    const unsigned short* bG = Bt + (size_t)(n0 + brow) * K + bcol;

    v8f acc[2][2] = {};

    for (int kk = 0; kk < K; kk += 32) {
        cp16_g2l(aG + kk,     &As[arow][acol]);
        cp16_g2l(bG + kk,     &Bs[brow][bcol]);
        cp16_g2l(bG + kk + 8, &Bs[brow][bcol + 8]);
        if (kk + 32 < K) {
            __builtin_prefetch(aG + kk + 32, 0, 1);
            __builtin_prefetch(bG + kk + 32, 0, 1);
        }
        cp_join();
        __syncthreads();

        union { v16bf v; v8bf h[2]; } a0, a1, b0, b1;
        {
            int ra  = wy * 32 + r;
            int off = hi ? 16 : 0;
            a0.h[0] = *(const v8bf*)&As[ra][off];
            a0.h[1] = *(const v8bf*)&As[ra][8 + off];
            a1.h[0] = *(const v8bf*)&As[ra + 16][off];
            a1.h[1] = *(const v8bf*)&As[ra + 16][8 + off];
        }
        {
            int rb  = wx * 32 + r;
            int off = hi ? 16 : 0;
            b0.h[0] = *(const v8bf*)&Bs[rb][off];
            b0.h[1] = *(const v8bf*)&Bs[rb][off + 8];
            b1.h[0] = *(const v8bf*)&Bs[rb + 16][off];
            b1.h[1] = *(const v8bf*)&Bs[rb + 16][off + 8];
        }
        acc[0][0] = bwmma(a0.v, b0.v, acc[0][0]);
        acc[0][1] = bwmma(a0.v, b1.v, acc[0][1]);
        acc[1][0] = bwmma(a1.v, b0.v, acc[1][0]);
        acc[1][1] = bwmma(a1.v, b1.v, acc[1][1]);
        __syncthreads();
    }

#pragma unroll
    for (int ty = 0; ty < 2; ++ty) {
#pragma unroll
        for (int tx = 0; tx < 2; ++tx) {
            const int col     = n0 + wx * 32 + tx * 16 + r;
            const int rowbase = m0 + wy * 32 + ty * 16 + (hi ? 8 : 0);
            const float bvv   = bias[col];
#pragma unroll
            for (int j = 0; j < 8; ++j)
                C[(size_t)(rowbase + j) * N + col] = acc[ty][tx][j] + bvv;
        }
    }
}

// ---------------------------------------------------------------------------
// Score kernel: s[b,h] = sum_q [ 1 - (sum_{masked k} p)/Z + n_masked*1e-20 ]
// grid (B*H, S/128), 8 waves; wave owns 16 q rows; 2 WMMAs per 16x16 k-tile.
// ---------------------------------------------------------------------------
__global__ __launch_bounds__(256)
void attn_score_kernel(const unsigned short* __restrict__ Qb,   // 8192x512 bf16
                       const unsigned short* __restrict__ Kb,   // 8192x512 bf16
                       const int* __restrict__ mask,            // B*S*S
                       float* __restrict__ s) {                 // B*H
    const int bh   = blockIdx.x;
    const int b    = bh >> 3;
    const int h    = bh & 7;
    const int wave = threadIdx.x >> 5;
    const int lane = threadIdx.x & 31;
    const int r    = lane & 15;
    const int hi   = (lane >> 4) & 1;
    const int q0   = blockIdx.y * 128 + wave * 16;

    // Q fragments: A-matrix layout, Dh split into two K=32 halves
    union { v16bf v; v8bf h2[2]; } aq0, aq1;
    {
        const unsigned short* p = Qb + (size_t)(b * SS + q0 + r) * DD + h * DH;
        int off = hi ? 16 : 0;
        aq0.h2[0] = *(const v8bf*)(p + off);
        aq0.h2[1] = *(const v8bf*)(p + 8 + off);
        aq1.h2[0] = *(const v8bf*)(p + 32 + off);
        aq1.h2[1] = *(const v8bf*)(p + 40 + off);
    }

    float m[8], Z[8], Ms[8], cn[8];
#pragma unroll
    for (int j = 0; j < 8; ++j) { m[j] = -INFINITY; Z[j] = 0.f; Ms[j] = 0.f; cn[j] = 0.f; }

    const int* mbase = mask + (size_t)b * SS * SS;

    for (int kt = 0; kt < SS; kt += 16) {
        // K^T fragments: B-matrix layout -> contiguous 16-element loads per lane
        const unsigned short* kp = Kb + (size_t)(b * SS + kt + r) * DD + h * DH;
        v16bf b0 = *(const v16bf*)(kp + (hi ? 16 : 0));
        v16bf b1 = *(const v16bf*)(kp + 32 + (hi ? 16 : 0));
        if (kt + 16 < SS) {
            __builtin_prefetch(kp + 16 * DD, 0, 1);
            __builtin_prefetch(mbase + (size_t)(q0 + (hi ? 8 : 0)) * SS + kt + 16, 0, 1);
        }

        v8f c = {};
        c = bwmma(aq0.v, b0, c);
        c = bwmma(aq1.v, b1, c);

        // online softmax statistics, row-wise (row spread over 16 lanes)
#pragma unroll
        for (int j = 0; j < 8; ++j) {
            int qrow = q0 + (hi ? 8 : 0) + j;
            float sc = c[j] * SCALE;
            float tmax = sc;
#pragma unroll
            for (int o = 8; o >= 1; o >>= 1) tmax = fmaxf(tmax, __shfl_xor(tmax, o, 16));
            float nm    = fmaxf(m[j], tmax);
            float alpha = __expf(m[j] - nm);
            float e     = __expf(sc - nm);
            int   mv    = mbase[(size_t)qrow * SS + kt + r];
            float tz = e;
            float tm = (mv == 1) ? e : 0.0f;
            float tc = (mv == 1) ? 1.0f : 0.0f;
#pragma unroll
            for (int o = 8; o >= 1; o >>= 1) {
                tz += __shfl_xor(tz, o, 16);
                tm += __shfl_xor(tm, o, 16);
                tc += __shfl_xor(tc, o, 16);
            }
            Z[j]  = Z[j] * alpha + tz;
            Ms[j] = Ms[j] * alpha + tm;
            cn[j] += tc;
            m[j]  = nm;
        }
    }

    float acc = 0.0f;
#pragma unroll
    for (int j = 0; j < 8; ++j) acc += 1.0f - Ms[j] / Z[j] + cn[j] * 1e-20f;
    acc *= (1.0f / 16.0f);                 // stats replicated across 16 lanes
#pragma unroll
    for (int o = 16; o >= 1; o >>= 1) acc += __shfl_xor(acc, o, 32);
    if (lane == 0) atomicAdd(&s[bh], acc);
}

// ---------------------------------------------------------------------------
// Host-side orchestration (graph-capture safe: launches only)
// ---------------------------------------------------------------------------
extern "C" void kernel_launch(void* const* d_in, const int* in_sizes, int n_in,
                              void* d_out, int out_size, void* d_ws, size_t ws_size,
                              hipStream_t stream) {
    const float* x    = (const float*)d_in[0];
    const int*   mask = (const int*)d_in[1];
    const float* Wq   = (const float*)d_in[2];
    const float* bq   = (const float*)d_in[3];
    const float* Wk   = (const float*)d_in[4];
    const float* bk   = (const float*)d_in[5];
    const float* Wv   = (const float*)d_in[6];
    const float* bv   = (const float*)d_in[7];
    const float* Wo   = (const float*)d_in[8];
    const float* bo   = (const float*)d_in[9];
    float* out = (float*)d_out;
    char*  ws  = (char*)d_ws;

    // workspace layout (256B-aligned)
    const size_t OFF_XBF  = 0;                                    // 8192x512 bf16
    const size_t OFF_WQKV = OFF_XBF  + (size_t)MROWS * DD * 2;    // 1536x512 bf16 (Wqkv^T)
    const size_t OFF_WO   = OFF_WQKV + (size_t)DD * NQKV * 2;     // 512x512 bf16 (Wo^T)
    const size_t OFF_BQKV = OFF_WO   + (size_t)DD * DD * 2;       // 1536 f32
    const size_t OFF_QKV  = OFF_BQKV + (size_t)NQKV * 4;          // 8192x1536 f32
    const size_t OFF_Q    = OFF_QKV  + (size_t)MROWS * NQKV * 4;  // 8192x512 bf16
    const size_t OFF_K    = OFF_Q    + (size_t)MROWS * DD * 2;    // 8192x512 bf16
    const size_t OFF_ATT  = OFF_K    + (size_t)MROWS * DD * 2;    // 8192x512 bf16
    const size_t OFF_S    = OFF_ATT  + (size_t)MROWS * DD * 2;    // 32 f32

    unsigned short* xbf   = (unsigned short*)(ws + OFF_XBF);
    unsigned short* wqkvT = (unsigned short*)(ws + OFF_WQKV);
    unsigned short* woT   = (unsigned short*)(ws + OFF_WO);
    float*          bqkv  = (float*)(ws + OFF_BQKV);
    float*          qkv   = (float*)(ws + OFF_QKV);
    unsigned short* qb    = (unsigned short*)(ws + OFF_Q);
    unsigned short* kb    = (unsigned short*)(ws + OFF_K);
    unsigned short* att   = (unsigned short*)(ws + OFF_ATT);
    float*          sbuf  = (float*)(ws + OFF_S);

    const int EW_X = (MROWS * DD + 255) / 256;   // 16384 blocks
    const int EW_W = (DD * DD + 255) / 256;      // 1024 blocks

    // 1) convert / pack operands to bf16 (weights transposed for the GEMMs)
    cvt_pack_kernel  <<<EW_X, 256, 0, stream>>>(x,  xbf, MROWS, DD);
    cvt_pack_t_kernel<<<EW_W, 256, 0, stream>>>(Wq, wqkvT, DD, DD, 0);
    cvt_pack_t_kernel<<<EW_W, 256, 0, stream>>>(Wk, wqkvT, DD, DD, DD);
    cvt_pack_t_kernel<<<EW_W, 256, 0, stream>>>(Wv, wqkvT, DD, DD, 2 * DD);
    cvt_pack_t_kernel<<<EW_W, 256, 0, stream>>>(Wo, woT,   DD, DD, 0);
    bias_pack_kernel <<<(NQKV + 255) / 256, 256, 0, stream>>>(bq, bk, bv, bqkv);

    // 2) fused QKV projection GEMM (WMMA bf16)
    gemm_bf16_kernel<<<dim3(MROWS / 64, NQKV / 128), 256, 0, stream>>>(
        xbf, wqkvT, bqkv, qkv, MROWS, NQKV, DD);

    // 3) Q,K -> bf16 for the score WMMAs
    qk_cvt_kernel<<<EW_X, 256, 0, stream>>>(qkv, qb, kb);

    // 4) per-(b,h) masked-softmax scalar reduction
    zero_s_kernel<<<1, 32, 0, stream>>>(sbuf);
    attn_score_kernel<<<dim3(BB * HH, SS / 128), 256, 0, stream>>>(qb, kb, mask, sbuf);

    // 5) att = bf16(v * s[b,h])
    scale_v_kernel<<<EW_X, 256, 0, stream>>>(qkv, sbuf, att);

    // 6) output projection GEMM (WMMA bf16) -> d_out
    gemm_bf16_kernel<<<dim3(MROWS / 64, DD / 128), 256, 0, stream>>>(
        att, woT, bo, out, MROWS, DD, DD);
}
